// Quantizer_28948079575589
// MI455X (gfx1250) — matrix operand order repaired and checked by reading, not compiled, not run on hardware
//
#include <hip/hip_runtime.h>
#include <stdint.h>

// ---------------------------------------------------------------------------
// Fused int4-quantize + low-rank outlier projection for MI455X (gfx1250).
//
// reference:
//   outlier = x @ W^T            (M=8192, N=64, K=4096)  -> v_wmma_f32_16x16x32_f16
//   scale   = fp16(rowabsmax/7)
//   q       = clip(rint(x/scale), -7, 7)
//
// Memory-bound: ~260 MiB HBM traffic -> ~11 us at 23.3 TB/s. Single fused
// pass over x computes absmax + WMMA together; quantize pass re-reads the
// 256 KB/WG tile out of L2.
// ---------------------------------------------------------------------------

typedef __attribute__((ext_vector_type(16))) _Float16 v16h;
typedef __attribute__((ext_vector_type(8)))  _Float16 v8h;
typedef __attribute__((ext_vector_type(8)))  float    v8f;

#define HIDDEN     4096
#define RANK       64
#define ROWS_TOTAL 8192          // 4 * 2048
#define TILE_ROWS  16
#define KCHUNK     32
#define NCHUNKS    (HIDDEN / KCHUNK)   // 128
#define THREADS    128                 // 4 waves (wave32)

__global__ __launch_bounds__(THREADS)
void fused_quant_lowrank(const float* __restrict__ x,
                         const float* __restrict__ W,
                         float* __restrict__ qOut,     // 8192*4096 (int values as f32)
                         float* __restrict__ sOut,     // 8192 (fp16 value widened to f32)
                         float* __restrict__ oOut)     // 8192*64
{
    __shared__ __align__(32) _Float16 aTile[TILE_ROWS][KCHUNK]; // 1 KB
    __shared__ __align__(32) _Float16 bTile[RANK][KCHUNK];      // 4 KB
    __shared__ float rowMax[TILE_ROWS];
    __shared__ float scaleLds[TILE_ROWS];

    const int t    = threadIdx.x;
    const int lane = t & 31;
    const int wave = t >> 5;                 // 0..3, one N-tile per wave
    const int row0 = blockIdx.x * TILE_ROWS;

    // x staging: 8 threads per row, float4 each -> 16 rows x 32 cols per chunk
    const int rLoc = t >> 3;                 // 0..15
    const int kSub = (t & 7) * 4;            // 0..28

    // W staging: 2 threads per rank row, 16 floats each
    const int wRow  = t >> 1;                // 0..63
    const int wkSub = (t & 1) * 16;          // 0 or 16

    // WMMA fragment coordinates (ISA 16-bit A 16x32 / B 32x16 layouts)
    const int m  = lane & 15;                // A row / B col
    const int h  = lane >> 4;                // lane-half selector
    const int n0 = wave * 16;                // this wave's rank slice

    const float* xRow  = x + (size_t)(row0 + rLoc) * HIDDEN;
    const float* wRowP = W + (size_t)wRow * HIDDEN;

    v8f   acc      = {};
    float localMax = 0.0f;

    for (int c = 0; c < NCHUNKS; ++c) {
        const int k0 = c * KCHUNK;

        // ---- stage A chunk (x) : absmax fused with the only HBM read of x
        float4 xv = *(const float4*)(xRow + k0 + kSub);
        localMax = fmaxf(localMax,
                         fmaxf(fmaxf(fabsf(xv.x), fabsf(xv.y)),
                               fmaxf(fabsf(xv.z), fabsf(xv.w))));
        aTile[rLoc][kSub + 0] = (_Float16)xv.x;
        aTile[rLoc][kSub + 1] = (_Float16)xv.y;
        aTile[rLoc][kSub + 2] = (_Float16)xv.z;
        aTile[rLoc][kSub + 3] = (_Float16)xv.w;

        // ---- stage B chunk (W) : 64x32, L2-resident after first WG touches it
        #pragma unroll
        for (int i = 0; i < 4; ++i) {
            float4 wv = *(const float4*)(wRowP + k0 + wkSub + i * 4);
            bTile[wRow][wkSub + i * 4 + 0] = (_Float16)wv.x;
            bTile[wRow][wkSub + i * 4 + 1] = (_Float16)wv.y;
            bTile[wRow][wkSub + i * 4 + 2] = (_Float16)wv.z;
            bTile[wRow][wkSub + i * 4 + 3] = (_Float16)wv.w;
        }

        // prefetch next x chunk while we compute (global_prefetch_b8)
        if (c + 1 < NCHUNKS)
            __builtin_prefetch(xRow + k0 + KCHUNK + kSub, 0, 1);

        __syncthreads();

        // ---- A fragment: lane {m,h}: halves 0..15 = K[h*16 .. h*16+15]
        v16h a = *(const v16h*)&aTile[m][h * 16];       // 32B aligned

        // ---- B fragment: lane n = m, K half selected by h; contiguous in bTile
        v16h b = *(const v16h*)&bTile[n0 + m][h * 16];  // 32B aligned

        // D = A x B + C  (16x16x32 f16 -> f32)
        acc = __builtin_amdgcn_wmma_f32_16x16x32_f16(
                  false, a, false, b, (short)0, acc, false, false);

        __syncthreads();
    }

    // ---- outlier writeback: C/D layout VGPR g -> M = g + 8*h, N = lane&15
    #pragma unroll
    for (int g = 0; g < 8; ++g) {
        const int mm = g + 8 * h;
        oOut[(size_t)(row0 + mm) * RANK + n0 + m] = acc[g];
    }

    // ---- per-row absmax: 8 consecutive lanes share a row (wave32)
    localMax = fmaxf(localMax, __shfl_xor(localMax, 1, 8));
    localMax = fmaxf(localMax, __shfl_xor(localMax, 2, 8));
    localMax = fmaxf(localMax, __shfl_xor(localMax, 4, 8));
    if ((t & 7) == 0) rowMax[rLoc] = localMax;
    __syncthreads();

    if (t < TILE_ROWS) {
        // match reference: scale computed in f32, cast to fp16, clip_ratio=1
        _Float16 sh = (_Float16)(rowMax[t] * (1.0f / 7.0f));
        sOut[row0 + t] = (float)sh;
        scaleLds[t]    = (float)sh;
    }
    __syncthreads();

    // ---- quantize pass: tile is L2-hot, so this re-read is cheap
    const float inv = 1.0f / scaleLds[rLoc];
    float* qRow = qOut + (size_t)(row0 + rLoc) * HIDDEN;
    for (int c = 0; c < NCHUNKS; ++c) {
        const int k = c * KCHUNK + kSub;
        float4 xv = *(const float4*)(xRow + k);
        float4 qv;
        qv.x = fminf(7.0f, fmaxf(-7.0f, rintf(xv.x * inv)));
        qv.y = fminf(7.0f, fmaxf(-7.0f, rintf(xv.y * inv)));
        qv.z = fminf(7.0f, fmaxf(-7.0f, rintf(xv.z * inv)));
        qv.w = fminf(7.0f, fmaxf(-7.0f, rintf(xv.w * inv)));
        *(float4*)(qRow + k) = qv;
    }
}

extern "C" void kernel_launch(void* const* d_in, const int* in_sizes, int n_in,
                              void* d_out, int out_size, void* d_ws, size_t ws_size,
                              hipStream_t stream) {
    (void)in_sizes; (void)n_in; (void)out_size; (void)d_ws; (void)ws_size;

    const float* x = (const float*)d_in[0];   // [4,2048,4096] f32
    const float* W = (const float*)d_in[1];   // [64,4096] f32

    // Outputs concatenated flat in return order, as float32:
    //   quantized_x : ROWS_TOTAL*HIDDEN
    //   scales      : ROWS_TOTAL
    //   outlier_x   : ROWS_TOTAL*RANK
    float* qOut = (float*)d_out;
    float* sOut = qOut + (size_t)ROWS_TOTAL * HIDDEN;
    float* oOut = sOut + ROWS_TOTAL;

    fused_quant_lowrank<<<ROWS_TOTAL / TILE_ROWS, THREADS, 0, stream>>>(
        x, W, qOut, sOut, oOut);
}
